// TopKPooling_37477884625086
// MI455X (gfx1250) — matrix operand order repaired and checked by reading, not compiled, not run on hardware
//
#include <hip/hip_runtime.h>
#include <hip/hip_bf16.h>

typedef __attribute__((ext_vector_type(16))) _Float16 v16h;
typedef __attribute__((ext_vector_type(8)))  float    v8f;

// -----------------------------------------------------------------------------
// Kernel 1: score[i] = tanh( dot(x[i,:], w) / ||w|| )  via V_WMMA_F32_16X16X32_F16
// Each wave (32 lanes) handles 16 nodes. C fixed to 256 (8 K-chunks of 32).
// B matrix = w broadcast to every column, so every column of D holds the dots.
// -----------------------------------------------------------------------------
__global__ __launch_bounds__(256)
void score_wmma_kernel(const float* __restrict__ x,
                       const float* __restrict__ w,
                       float* __restrict__ score,
                       int n_total)
{
    const int lane = threadIdx.x & 31;
    const int wave = threadIdx.x >> 5;
    const int node_base = (blockIdx.x * 8 + wave) * 16;
    if (node_base + 16 > n_total) return;            // wave-uniform guard

    const int m = lane & 15;       // A-matrix row handled by this lane
    const int h = lane >> 4;       // half-wave selector

    // ||w|| (uniform across lanes -> scalar loads)
    float nw = 0.f;
    #pragma unroll 8
    for (int k = 0; k < 256; ++k) { float t = w[k]; nw += t * t; }
    const float inv_norm = rsqrtf(nw);

    const float* __restrict__ xrow = x + (size_t)(node_base + m) * 256;

    v8f c = {};
    #pragma unroll
    for (int chunk = 0; chunk < 8; ++chunk) {
        const int k0 = chunk * 32;
        v16h a, b;
        // A: 16x32 f16 layout — halves 0..7: K = 8h+j ; halves 8..15: K = 16+8h+j
        #pragma unroll
        for (int j = 0; j < 8; ++j) {
            a[j]     = (_Float16)xrow[k0 + 8 * h + j];
            a[8 + j] = (_Float16)xrow[k0 + 16 + 8 * h + j];
        }
        // B: 32x16 f16 layout — halves 0..7: K = 8h+j ; halves 8..15: K = 16+8h+j (matches A).
        // Broadcast w across all columns -> value depends only on K.
        #pragma unroll
        for (int j = 0; j < 16; ++j) {
            b[j] = (_Float16)w[k0 + 8 * h + (j & 7) + 16 * (j >> 3)];
        }
        c = __builtin_amdgcn_wmma_f32_16x16x32_f16(
                false, a, false, b, (short)0, c, false, false);
    }

    // C/D layout: lanes 0..15 hold rows 0..7 in vgprs 0..7 (col = lane&15),
    // lanes 16..31 hold rows 8..15. Column 0 lanes (0 and 16) write results.
    if ((lane & 15) == 0) {
        #pragma unroll
        for (int r = 0; r < 8; ++r) {
            score[node_base + 8 * h + r] = tanhf(c[r] * inv_norm);
        }
    }
}

// -----------------------------------------------------------------------------
// Kernel 2: per-graph top-256-of-512 via bitonic sort in LDS.
// Descending score, ascending-index tie-break (matches jax.lax.top_k).
// Writes perm / score_perm (ws), new_id remap table (ws), and the small outputs.
// -----------------------------------------------------------------------------
__global__ __launch_bounds__(256)
void topk_kernel(const float* __restrict__ score,
                 int*   __restrict__ new_id,
                 int*   __restrict__ perm,
                 float* __restrict__ score_perm,
                 float* __restrict__ o_batch,
                 float* __restrict__ o_perm,
                 float* __restrict__ o_score)
{
    __shared__ float ss[512];
    __shared__ int   si[512];

    const int g = blockIdx.x;
    const int t = threadIdx.x;          // 256 threads, 2 elements each
    const int base = g * 512;

    ss[t]       = score[base + t];       si[t]       = t;
    ss[t + 256] = score[base + 256 + t]; si[t + 256] = 256 + t;
    __syncthreads();

    for (int len = 2; len <= 512; len <<= 1) {
        for (int j = len >> 1; j > 0; j >>= 1) {
            #pragma unroll
            for (int wv = 0; wv < 2; ++wv) {
                const int ii  = t + wv * 256;
                const int ixj = ii ^ j;
                if (ixj > ii) {
                    const bool  up = ((ii & len) == 0);
                    const float s1 = ss[ii],  s2 = ss[ixj];
                    const int   i1 = si[ii],  i2 = si[ixj];
                    // "element at ii comes after element at ixj" in descending order
                    const bool after = (s2 > s1) || (s2 == s1 && i2 < i1);
                    if (after == up) {
                        ss[ii] = s2; ss[ixj] = s1;
                        si[ii] = i2; si[ixj] = i1;
                    }
                }
            }
            __syncthreads();
        }
    }

    // First 256 entries = top-k in descending order.
    const int   gi  = base + si[t];
    const int   pos = g * 256 + t;
    const float sc  = ss[t];
    perm[pos]       = gi;
    score_perm[pos] = sc;
    new_id[gi]      = pos;                 // selected -> new index
    new_id[base + si[t + 256]] = -1;       // dropped  -> -1
    o_batch[pos] = (float)g;
    o_perm[pos]  = (float)gi;
    o_score[pos] = sc;
}

// -----------------------------------------------------------------------------
// Kernel 3: x_out[i,:] = x[perm[i],:] * score_perm[i]   (float4 vectorized)
// -----------------------------------------------------------------------------
__global__ __launch_bounds__(256)
void gather_scale_kernel(const float* __restrict__ x,
                         const int*   __restrict__ perm,
                         const float* __restrict__ score_perm,
                         float* __restrict__ o_x,
                         int n_sel, int c4)   // c4 = C/4
{
    const int i = blockIdx.x * blockDim.x + threadIdx.x;
    const int total = n_sel * c4;
    if (i >= total) return;
    const int node = i / c4;
    const int q    = i - node * c4;
    const int src  = perm[node];
    const float sc = score_perm[node];
    const float4* __restrict__ xs = (const float4*)x;
    float4 v = xs[(size_t)src * c4 + q];
    v.x *= sc; v.y *= sc; v.z *= sc; v.w *= sc;
    ((float4*)o_x)[(size_t)node * c4 + q] = v;
}

// -----------------------------------------------------------------------------
// Kernel 4: filter_adj — keep edges with both endpoints selected, remap ids.
// -----------------------------------------------------------------------------
__global__ __launch_bounds__(256)
void edge_kernel(const int* __restrict__ edge_index,   // [2, E] row-major
                 const int* __restrict__ new_id,
                 float* __restrict__ o_ei,              // [2, E] as float
                 float* __restrict__ o_keep,            // [E] as float
                 int E)
{
    const int e = blockIdx.x * blockDim.x + threadIdx.x;
    if (e >= E) return;
    const int row = edge_index[e];
    const int col = edge_index[E + e];
    const int nr = new_id[row];
    const int nc = new_id[col];
    const bool keep = (nr >= 0) && (nc >= 0);
    o_ei[e]     = (float)(keep ? nr : -1);
    o_ei[E + e] = (float)(keep ? nc : -1);
    o_keep[e]   = keep ? 1.0f : 0.0f;
}

// -----------------------------------------------------------------------------
extern "C" void kernel_launch(void* const* d_in, const int* in_sizes, int n_in,
                              void* d_out, int out_size, void* d_ws, size_t ws_size,
                              hipStream_t stream)
{
    const float* x          = (const float*)d_in[0];
    const float* w          = (const float*)d_in[1];
    const int*   edge_index = (const int*)d_in[2];
    // d_in[3] = batch (unused: batch_out derived), d_in[4] = num_graphs (scalar)

    const int C       = in_sizes[1];          // 256
    const int n_total = in_sizes[0] / C;      // 65536
    const int E       = in_sizes[2] / 2;      // 2097152
    const int N_PER   = 512;                  // reference constant
    const int G       = n_total / N_PER;      // 128
    const int K       = N_PER / 2;            // ceil(0.5*512) = 256
    const int n_sel   = G * K;                // 32768

    // workspace layout
    float* ws_score  = (float*)d_ws;                  // [n_total]
    int*   ws_newid  = (int*)(ws_score + n_total);    // [n_total]
    int*   ws_perm   = (int*)(ws_newid + n_total);    // [n_sel]
    float* ws_scperm = (float*)(ws_perm + n_sel);     // [n_sel]

    // output layout (floats, concatenated in reference return order)
    float* o_x     = (float*)d_out;                   // n_sel*C
    float* o_ei    = o_x    + (size_t)n_sel * C;      // 2*E
    float* o_keep  = o_ei   + 2ull * (size_t)E;       // E
    float* o_batch = o_keep + (size_t)E;              // n_sel
    float* o_perm  = o_batch + n_sel;                 // n_sel
    float* o_score = o_perm  + n_sel;                 // n_sel

    // 1) scores via WMMA (each block: 8 waves x 16 nodes = 128 nodes)
    score_wmma_kernel<<<n_total / 128, 256, 0, stream>>>(x, w, ws_score, n_total);

    // 2) per-graph top-k
    topk_kernel<<<G, 256, 0, stream>>>(ws_score, ws_newid, ws_perm, ws_scperm,
                                       o_batch, o_perm, o_score);

    // 3) gather + rescale
    const int c4 = C / 4;
    const int total3 = n_sel * c4;
    gather_scale_kernel<<<(total3 + 255) / 256, 256, 0, stream>>>(
        x, ws_perm, ws_scperm, o_x, n_sel, c4);

    // 4) edge filtering / remap
    edge_kernel<<<(E + 255) / 256, 256, 0, stream>>>(
        edge_index, ws_newid, o_ei, o_keep, E);
}
